// ActorGNN_40767829574200
// MI455X (gfx1250) — matrix operand order repaired and checked by reading, not compiled, not run on hardware
//
#include <hip/hip_runtime.h>
#include <hip/hip_bf16.h>
#include <math.h>

#define NN 50000
#define EE 800000
#define MM 10000
#define TT 50
#define E2 (EE + NN)

typedef _Float16 v16h __attribute__((ext_vector_type(16)));
typedef float    v8f  __attribute__((ext_vector_type(8)));

// ---------------- utility kernels ----------------
__global__ void k_fill_f32(float* p, float v, int n) {
    int i = blockIdx.x * blockDim.x + threadIdx.x;
    if (i < n) p[i] = v;
}
__global__ void k_fill_i32(int* p, int v, int n) {
    int i = blockIdx.x * blockDim.x + threadIdx.x;
    if (i < n) p[i] = v;
}
__global__ void k_scatter_addmap(const int* comp_idx, int* addmap, int m) {
    int i = blockIdx.x * blockDim.x + threadIdx.x;
    if (i < m) addmap[comp_idx[i]] = i;
}
// W: [Ncols][K] row-major fp32  ->  Wt: [K][Ncols] f16
__global__ void k_convert_wt(const float* W, _Float16* Wt, int K, int Ncols) {
    int i = blockIdx.x * blockDim.x + threadIdx.x;
    if (i < K * Ncols) {
        int k = i / Ncols, n = i % Ncols;
        Wt[i] = (_Float16)W[n * K + k];
    }
}

__device__ inline void atomicMaxF(float* addr, float v) {
    // monotonic float max via int/uint atomics (valid for all finite floats)
    if (v >= 0.f) atomicMax((int*)addr, __float_as_int(v));
    else          atomicMin((unsigned int*)addr, __float_as_uint(v));
}

// ---------------- GRU: one wave32 per component, lane = hidden unit ----------------
__global__ __launch_bounds__(128) void k_gru(const float* __restrict__ seq,
                                             const float* __restrict__ Wi,
                                             const float* __restrict__ bi,
                                             const float* __restrict__ Wh,
                                             const float* __restrict__ bh,
                                             float* __restrict__ Ht) {
    __shared__ float WhS[96 * 32];
    __shared__ float WiS[96 * 2];
    __shared__ float biS[96], bhS[96];
    for (int i = threadIdx.x; i < 96 * 32; i += 128) WhS[i] = Wh[i];
    for (int i = threadIdx.x; i < 192; i += 128)     WiS[i] = Wi[i];
    for (int i = threadIdx.x; i < 96; i += 128) { biS[i] = bi[i]; bhS[i] = bh[i]; }
    __syncthreads();
    int wave = threadIdx.x >> 5, lane = threadIdx.x & 31;
    int m = blockIdx.x * 4 + wave;
    if (m >= MM) return;
    const float* sm = seq + (size_t)m * TT * 2;
    float h = 0.f;
    for (int t = 0; t < TT; t++) {
        float x0 = sm[t * 2], x1 = sm[t * 2 + 1];
        float gir = x0 * WiS[lane * 2]        + x1 * WiS[lane * 2 + 1]        + biS[lane];
        float giz = x0 * WiS[(32 + lane) * 2] + x1 * WiS[(32 + lane) * 2 + 1] + biS[32 + lane];
        float gin = x0 * WiS[(64 + lane) * 2] + x1 * WiS[(64 + lane) * 2 + 1] + biS[64 + lane];
        float ghr = bhS[lane], ghz = bhS[32 + lane], ghn = bhS[64 + lane];
        #pragma unroll 8
        for (int j = 0; j < 32; j++) {
            float hj = __shfl(h, j, 32);
            ghr += hj * WhS[lane * 32 + j];
            ghz += hj * WhS[(32 + lane) * 32 + j];
            ghn += hj * WhS[(64 + lane) * 32 + j];
        }
        float r = 1.f / (1.f + expf(-(gir + ghr)));
        float z = 1.f / (1.f + expf(-(giz + ghz)));
        float nn = tanhf(gin + r * ghn);
        h = (1.f - z) * nn + z * h;
    }
    Ht[m * 32 + lane] = h;
}

// ---------------- node features + lin_in (56 -> 64, ReLU) ----------------
__global__ __launch_bounds__(256) void k_feat(const float* __restrict__ x,
                                              const int* __restrict__ ntype,
                                              const float* __restrict__ temb,
                                              const int* __restrict__ addmap,
                                              const float* __restrict__ Ht,
                                              const float* __restrict__ W,
                                              const float* __restrict__ b,
                                              float* __restrict__ out32,
                                              _Float16* __restrict__ out16) {
    __shared__ float WS[64 * 56];
    __shared__ float bS[64];
    __shared__ float inS[4][56];
    for (int i = threadIdx.x; i < 64 * 56; i += 256) WS[i] = W[i];
    if (threadIdx.x < 64) bS[threadIdx.x] = b[threadIdx.x];
    int g = threadIdx.x >> 6;   // node within block
    int o = threadIdx.x & 63;   // output channel
    int node = blockIdx.x * 4 + g;
    if (node < NN && o < 56) {
        float v;
        if (o < 16)      v = x[node * 16 + o];
        else if (o < 48) { int am = addmap[node]; v = (am >= 0) ? Ht[am * 32 + (o - 16)] : 0.f; }
        else             v = temb[ntype[node] * 8 + (o - 48)];
        inS[g][o] = v;
    }
    __syncthreads();
    if (node < NN) {
        float s = bS[o];
        #pragma unroll 8
        for (int k = 0; k < 56; k++) s += inS[g][k] * WS[o * 56 + k];
        s = fmaxf(s, 0.f);
        out32[node * 64 + o] = s;
        out16[node * 64 + o] = (_Float16)s;
    }
}

// ---------------- self-loop edge_attr (fill_value='mean') ----------------
__global__ void k_loop_acc(const int* __restrict__ ei, const float* __restrict__ ea,
                           float* __restrict__ deg, float* __restrict__ lsum) {
    int i = blockIdx.x * blockDim.x + threadIdx.x;
    if (i < EE * 8) {
        int e = i >> 3, j = i & 7;
        int dst = ei[EE + e];
        atomicAdd(&lsum[dst * 8 + j], ea[i]);
        if (j == 0) atomicAdd(&deg[dst], 1.f);
    }
}
__global__ void k_loop_fin(float* lsum, const float* deg) {
    int i = blockIdx.x * blockDim.x + threadIdx.x;
    if (i < NN * 8) lsum[i] /= fmaxf(deg[i >> 3], 1.f);
}

// ---------------- WMMA GEMM: C[N,128] = A[N,64](f16) * Wt[64,128](f16) + bias ----------------
// persistent: wave w of each block owns column-tile w; B-fragments hoisted, grid-stride over
// row tiles so the weight fragments are loaded once per wave and A streams through WMMA.
__global__ __launch_bounds__(256) void k_gemm_wmma(const _Float16* __restrict__ A,
                                                   const _Float16* __restrict__ Wt,
                                                   const float* __restrict__ bias,
                                                   float* __restrict__ C, int nrows) {
    int wave = threadIdx.x >> 5, lane = threadIdx.x & 31;
    int half = lane >> 4, l15 = lane & 15;
    int col = wave * 16 + l15;
    // B-frag 32x16 f16: lanes 0-15 hold K = k0+0..15, lanes 16-31 hold K = k0+16..31 (col = N)
    v16h b0, b1;
    #pragma unroll
    for (int i = 0; i < 16; i++) b0[i] = Wt[(half * 16 + i) * 128 + col];
    #pragma unroll
    for (int i = 0; i < 16; i++) b1[i] = Wt[(32 + half * 16 + i) * 128 + col];
    float bcol = bias[col];
    int ntiles = (nrows + 15) >> 4;
    int stride = gridDim.x;
    for (int mt = blockIdx.x; mt < ntiles; mt += stride) {
        int mbase = mt * 16;
        // A-frag 16x32 f16: lanes 0-15 K = k0+0..15; lanes 16-31 K = k0+16..31
        const _Float16* arow = A + (size_t)(mbase + l15) * 64 + half * 16;
        // prefetch next row tile for this wave (global_prefetch_b8)
        if (mbase + stride * 16 < nrows)
            __builtin_prefetch(arow + (size_t)stride * 16 * 64, 0, 1);
        v16h a0, a1;
        #pragma unroll
        for (int i = 0; i < 8; i++) { a0[i] = arow[i];      a0[8 + i] = arow[8 + i]; }
        #pragma unroll
        for (int i = 0; i < 8; i++) { a1[i] = arow[32 + i]; a1[8 + i] = arow[40 + i]; }
        v8f c = {};
        c = __builtin_amdgcn_wmma_f32_16x16x32_f16(false, a0, false, b0, (short)0, c, false, false);
        c = __builtin_amdgcn_wmma_f32_16x16x32_f16(false, a1, false, b1, (short)0, c, false, false);
        #pragma unroll
        for (int v = 0; v < 8; v++) {
            int row = mbase + v + half * 8;
            C[(size_t)row * 128 + col] = c[v] + bcol;
        }
    }
}

// ---------------- GATv2 edge scoring (wave per edge) ----------------
__global__ __launch_bounds__(256) void k_edge_logits(const int* __restrict__ ei,
                                                     const float* __restrict__ ea,
                                                     const float* __restrict__ lattr,
                                                     const float* __restrict__ XL,
                                                     const float* __restrict__ XR,
                                                     const float* __restrict__ We,
                                                     const float* __restrict__ att,
                                                     float* __restrict__ logits,
                                                     float* __restrict__ mx) {
    __shared__ float WeS[128 * 8];
    __shared__ float attS[128];
    for (int i = threadIdx.x; i < 1024; i += 256) WeS[i] = We[i];
    if (threadIdx.x < 128) attS[threadIdx.x] = att[threadIdx.x];
    __syncthreads();
    int wave = threadIdx.x >> 5, lane = threadIdx.x & 31;
    int e = blockIdx.x * 8 + wave;
    if (e >= E2) return;
    int src, dst; const float* eav;
    if (e < EE) { src = ei[e]; dst = ei[EE + e]; eav = ea + (size_t)e * 8; }
    else        { src = dst = e - EE;            eav = lattr + (size_t)src * 8; }
    float ev[8];
    #pragma unroll
    for (int j = 0; j < 8; j++) ev[j] = eav[j];
    const float* xls = XL + (size_t)src * 128;
    const float* xrd = XR + (size_t)dst * 128;
    float p0 = 0.f, p1 = 0.f;
    #pragma unroll
    for (int dd = 0; dd < 2; dd++) {
        int d = dd * 32 + lane;
        float e0 = 0.f, e1 = 0.f;
        #pragma unroll
        for (int j = 0; j < 8; j++) { e0 += ev[j] * WeS[d * 8 + j]; e1 += ev[j] * WeS[(64 + d) * 8 + j]; }
        float s0 = xls[d] + xrd[d] + e0;              s0 = (s0 > 0.f) ? s0 : 0.2f * s0;
        float s1 = xls[64 + d] + xrd[64 + d] + e1;    s1 = (s1 > 0.f) ? s1 : 0.2f * s1;
        p0 += attS[d] * s0;  p1 += attS[64 + d] * s1;
    }
    for (int off = 16; off; off >>= 1) { p0 += __shfl_down(p0, off, 32); p1 += __shfl_down(p1, off, 32); }
    if (lane == 0) {
        logits[(size_t)e * 2]     = p0;
        logits[(size_t)e * 2 + 1] = p1;
        atomicMaxF(&mx[dst * 2],     p0);
        atomicMaxF(&mx[dst * 2 + 1], p1);
    }
}

__global__ void k_edge_exp(const int* __restrict__ ei, float* __restrict__ logits,
                           const float* __restrict__ mx, float* __restrict__ den) {
    int i = blockIdx.x * blockDim.x + threadIdx.x;
    if (i >= E2 * 2) return;
    int e = i >> 1, h = i & 1;
    int dst = (e < EE) ? ei[EE + e] : (e - EE);
    float ex = expf(logits[i] - mx[dst * 2 + h]);
    logits[i] = ex;
    atomicAdd(&den[dst * 2 + h], ex);
}

__global__ __launch_bounds__(256) void k_edge_agg(const int* __restrict__ ei,
                                                  const float* __restrict__ logits,
                                                  const float* __restrict__ den,
                                                  const float* __restrict__ XL,
                                                  float* __restrict__ acc) {
    int wave = threadIdx.x >> 5, lane = threadIdx.x & 31;
    int e = blockIdx.x * 8 + wave;
    if (e >= E2) return;
    int src, dst;
    if (e < EE) { src = ei[e]; dst = ei[EE + e]; } else { src = dst = e - EE; }
    float a0 = logits[(size_t)e * 2]     / den[dst * 2];
    float a1 = logits[(size_t)e * 2 + 1] / den[dst * 2 + 1];
    const float* xls = XL + (size_t)src * 128;
    float* ap = acc + (size_t)dst * 128;
    #pragma unroll
    for (int q = 0; q < 4; q++) {
        int i = q * 32 + lane;
        float al = (i < 64) ? a0 : a1;
        atomicAdd(&ap[i], xls[i] * al);
    }
}

__global__ void k_finalize(const float* __restrict__ acc, const float* __restrict__ bias,
                           float* __restrict__ out32, _Float16* __restrict__ out16, int elu) {
    int i = blockIdx.x * blockDim.x + threadIdx.x;
    if (i >= NN * 64) return;
    int node = i >> 6, d = i & 63;
    float v = 0.5f * (acc[(size_t)node * 128 + d] + acc[(size_t)node * 128 + 64 + d]) + bias[d];
    if (elu) v = (v > 0.f) ? v : expm1f(v);
    out32[i] = v;
    out16[i] = (_Float16)v;
}

// ---------------- graph readout ----------------
__global__ __launch_bounds__(256) void k_colsum(const float* __restrict__ H, float* __restrict__ hsum) {
    __shared__ float red[256];
    int col = threadIdx.x & 63, q = threadIdx.x >> 6;
    float s = 0.f;
    for (int r = blockIdx.x * 4 + q; r < NN; r += gridDim.x * 4) s += H[(size_t)r * 64 + col];
    red[threadIdx.x] = s;
    __syncthreads();
    if (q == 0) atomicAdd(&hsum[col], red[col] + red[64 + col] + red[128 + col] + red[192 + col]);
}

__global__ __launch_bounds__(64) void k_graph_head(const float* __restrict__ hsum,
                                                   const float* __restrict__ lgW, const float* __restrict__ lgb,
                                                   const float* __restrict__ th1W, const float* __restrict__ th1b,
                                                   const float* __restrict__ th2W, const float* __restrict__ th2b,
                                                   const float* __restrict__ la1W,
                                                   const float* __restrict__ lstheta,
                                                   float* __restrict__ gpart, float* __restrict__ out) {
    __shared__ float gS[64], t1S[64];
    int o = threadIdx.x;
    float s = lgb[o];
    for (int k = 0; k < 64; k++) s += (hsum[k] * (1.f / NN)) * lgW[o * 64 + k];
    gS[o] = tanhf(s);
    __syncthreads();
    float t = th1b[o], gp = 0.f;
    for (int k = 0; k < 64; k++) { t += gS[k] * th1W[o * 64 + k]; gp += gS[k] * la1W[o * 128 + 64 + k]; }
    t1S[o] = fmaxf(t, 0.f);
    gpart[o] = gp;
    __syncthreads();
    if (o < 6) {
        float mt = th2b[o];
        for (int k = 0; k < 64; k++) mt += t1S[k] * th2W[o * 64 + k];
        out[2 * MM + o]     = mt;
        out[2 * MM + 6 + o] = lstheta[o];
    }
}

__global__ __launch_bounds__(256) void k_lambda(const float* __restrict__ H, const int* __restrict__ comp_idx,
                                                const float* __restrict__ la1W, const float* __restrict__ la1b,
                                                const float* __restrict__ gpart, const float* __restrict__ la2W,
                                                const float* __restrict__ la2b, const float* __restrict__ lsl,
                                                float* __restrict__ out) {
    __shared__ float AS[64 * 64];
    __shared__ float bS[64], gpS[64], wS[64];
    for (int i = threadIdx.x; i < 4096; i += 256) { int j = i >> 6, k = i & 63; AS[i] = la1W[j * 128 + k]; }
    if (threadIdx.x < 64) { bS[threadIdx.x] = la1b[threadIdx.x]; gpS[threadIdx.x] = gpart[threadIdx.x]; wS[threadIdx.x] = la2W[threadIdx.x]; }
    __syncthreads();
    int m = blockIdx.x * 256 + threadIdx.x;
    if (m >= MM) return;
    const float* hc = H + (size_t)comp_idx[m] * 64;
    float hr[64];
    #pragma unroll
    for (int k = 0; k < 64; k++) hr[k] = hc[k];
    float mu = la2b[0];
    for (int j = 0; j < 64; j++) {
        float s = bS[j] + gpS[j];
        #pragma unroll 8
        for (int k = 0; k < 64; k++) s += hr[k] * AS[j * 64 + k];
        mu += fmaxf(s, 0.f) * wS[j];
    }
    out[m]      = mu;
    out[MM + m] = lsl[0];
}

// ---------------- launch ----------------
extern "C" void kernel_launch(void* const* d_in, const int* in_sizes, int n_in,
                              void* d_out, int out_size, void* d_ws, size_t ws_size,
                              hipStream_t stream) {
    // inputs (setup_inputs insertion order; params flattened in make_params insertion order)
    const float* x        = (const float*)d_in[0];
    const int*   ntype    = (const int*)  d_in[1];
    const int*   ei       = (const int*)  d_in[2];   // [2,E]
    const float* ea       = (const float*)d_in[3];   // [E,8]
    const float* seq      = (const float*)d_in[4];   // [M,T,2]
    const int*   comp_idx = (const int*)  d_in[5];
    const float* temb     = (const float*)d_in[6];
    const float* lin_in_W = (const float*)d_in[7];
    const float* lin_in_b = (const float*)d_in[8];
    const float* g1_Wl = (const float*)d_in[9],  *g1_bl = (const float*)d_in[10];
    const float* g1_Wr = (const float*)d_in[11], *g1_br = (const float*)d_in[12];
    const float* g1_We = (const float*)d_in[13], *g1_att = (const float*)d_in[14], *g1_bias = (const float*)d_in[15];
    const float* g2_Wl = (const float*)d_in[16], *g2_bl = (const float*)d_in[17];
    const float* g2_Wr = (const float*)d_in[18], *g2_br = (const float*)d_in[19];
    const float* g2_We = (const float*)d_in[20], *g2_att = (const float*)d_in[21], *g2_bias = (const float*)d_in[22];
    const float* lgW = (const float*)d_in[23], *lgb = (const float*)d_in[24];
    const float* gru_Wi = (const float*)d_in[25], *gru_bi = (const float*)d_in[26];
    const float* gru_Wh = (const float*)d_in[27], *gru_bh = (const float*)d_in[28];
    const float* la1W = (const float*)d_in[29], *la1b = (const float*)d_in[30];
    const float* la2W = (const float*)d_in[31], *la2b = (const float*)d_in[32];
    const float* th1W = (const float*)d_in[33], *th1b = (const float*)d_in[34];
    const float* th2W = (const float*)d_in[35], *th2b = (const float*)d_in[36];
    const float* lsl = (const float*)d_in[37], *lstheta = (const float*)d_in[38];
    float* out = (float*)d_out;

    // workspace carve (256B aligned)
    char* ws = (char*)d_ws;
    auto carve = [&](size_t bytes) -> void* {
        void* p = ws; ws += (bytes + 255) & ~(size_t)255; return p;
    };
    float*    Ht     = (float*)   carve((size_t)MM * 32 * 4);
    int*      addmap = (int*)     carve((size_t)NN * 4);
    float*    F32    = (float*)   carve((size_t)NN * 64 * 4);
    _Float16* F16    = (_Float16*)carve((size_t)NN * 64 * 2);
    float*    XL     = (float*)   carve((size_t)NN * 128 * 4);
    float*    XR     = (float*)   carve((size_t)NN * 128 * 4);
    float*    ACC    = (float*)   carve((size_t)NN * 128 * 4);
    float*    HOUT   = (float*)   carve((size_t)NN * 64 * 4);
    float*    logits = (float*)   carve((size_t)E2 * 2 * 4);
    float*    mx     = (float*)   carve((size_t)NN * 2 * 4);
    float*    den    = (float*)   carve((size_t)NN * 2 * 4);
    float*    deg    = (float*)   carve((size_t)NN * 4);
    float*    lsum   = (float*)   carve((size_t)NN * 8 * 4);
    _Float16* W1l    = (_Float16*)carve(64 * 128 * 2);
    _Float16* W1r    = (_Float16*)carve(64 * 128 * 2);
    _Float16* W2l    = (_Float16*)carve(64 * 128 * 2);
    _Float16* W2r    = (_Float16*)carve(64 * 128 * 2);
    float*    hsum   = (float*)   carve(64 * 4);
    float*    gpart  = (float*)   carve(64 * 4);

    auto cdiv = [](int a, int b) { return (a + b - 1) / b; };

    // prep
    k_fill_i32<<<cdiv(NN, 256), 256, 0, stream>>>(addmap, -1, NN);
    k_scatter_addmap<<<cdiv(MM, 256), 256, 0, stream>>>(comp_idx, addmap, MM);
    k_convert_wt<<<cdiv(8192, 256), 256, 0, stream>>>(g1_Wl, W1l, 64, 128);
    k_convert_wt<<<cdiv(8192, 256), 256, 0, stream>>>(g1_Wr, W1r, 64, 128);
    k_convert_wt<<<cdiv(8192, 256), 256, 0, stream>>>(g2_Wl, W2l, 64, 128);
    k_convert_wt<<<cdiv(8192, 256), 256, 0, stream>>>(g2_Wr, W2r, 64, 128);

    // GRU + node features
    k_gru<<<cdiv(MM, 4), 128, 0, stream>>>(seq, gru_Wi, gru_bi, gru_Wh, gru_bh, Ht);
    k_feat<<<cdiv(NN, 4), 256, 0, stream>>>(x, ntype, temb, addmap, Ht, lin_in_W, lin_in_b, F32, F16);

    // self-loop edge_attr (mean of incoming)
    k_fill_f32<<<cdiv(NN, 256), 256, 0, stream>>>(deg, 0.f, NN);
    k_fill_f32<<<cdiv(NN * 8, 256), 256, 0, stream>>>(lsum, 0.f, NN * 8);
    k_loop_acc<<<cdiv(EE * 8, 256), 256, 0, stream>>>(ei, ea, deg, lsum);
    k_loop_fin<<<cdiv(NN * 8, 256), 256, 0, stream>>>(lsum, deg);

    const int gemm_grid = 512;               // persistent blocks; ~6 row tiles per wave
    const int edge_grid = cdiv(E2, 8);

    // ---- GAT layer 1 ----
    k_gemm_wmma<<<gemm_grid, 256, 0, stream>>>(F16, W1l, g1_bl, XL, NN);
    k_gemm_wmma<<<gemm_grid, 256, 0, stream>>>(F16, W1r, g1_br, XR, NN);
    k_fill_f32<<<cdiv(NN * 2, 256), 256, 0, stream>>>(mx, -INFINITY, NN * 2);
    k_fill_f32<<<cdiv(NN * 2, 256), 256, 0, stream>>>(den, 0.f, NN * 2);
    k_fill_f32<<<cdiv(NN * 128, 256), 256, 0, stream>>>(ACC, 0.f, NN * 128);
    k_edge_logits<<<edge_grid, 256, 0, stream>>>(ei, ea, lsum, XL, XR, g1_We, g1_att, logits, mx);
    k_edge_exp<<<cdiv(E2 * 2, 256), 256, 0, stream>>>(ei, logits, mx, den);
    k_edge_agg<<<edge_grid, 256, 0, stream>>>(ei, logits, den, XL, ACC);
    k_finalize<<<cdiv(NN * 64, 256), 256, 0, stream>>>(ACC, g1_bias, F32, F16, 1); // ELU

    // ---- GAT layer 2 ----
    k_gemm_wmma<<<gemm_grid, 256, 0, stream>>>(F16, W2l, g2_bl, XL, NN);
    k_gemm_wmma<<<gemm_grid, 256, 0, stream>>>(F16, W2r, g2_br, XR, NN);
    k_fill_f32<<<cdiv(NN * 2, 256), 256, 0, stream>>>(mx, -INFINITY, NN * 2);
    k_fill_f32<<<cdiv(NN * 2, 256), 256, 0, stream>>>(den, 0.f, NN * 2);
    k_fill_f32<<<cdiv(NN * 128, 256), 256, 0, stream>>>(ACC, 0.f, NN * 128);
    k_edge_logits<<<edge_grid, 256, 0, stream>>>(ei, ea, lsum, XL, XR, g2_We, g2_att, logits, mx);
    k_edge_exp<<<cdiv(E2 * 2, 256), 256, 0, stream>>>(ei, logits, mx, den);
    k_edge_agg<<<edge_grid, 256, 0, stream>>>(ei, logits, den, XL, ACC);
    k_finalize<<<cdiv(NN * 64, 256), 256, 0, stream>>>(ACC, g2_bias, HOUT, F16, 0);

    // ---- readout + heads ----
    k_fill_f32<<<1, 64, 0, stream>>>(hsum, 0.f, 64);
    k_colsum<<<128, 256, 0, stream>>>(HOUT, hsum);
    k_graph_head<<<1, 64, 0, stream>>>(hsum, lgW, lgb, th1W, th1b, th2W, th2b, la1W, lstheta, gpart, out);
    k_lambda<<<cdiv(MM, 256), 256, 0, stream>>>(HOUT, comp_idx, la1W, la1b, gpart, la2W, la2b, lsl, out);
}
